// TemporalSheafTransport_11510512353864
// MI455X (gfx1250) — matrix operand order repaired and checked by reading, not compiled, not run on hardware
//
#include <hip/hip_runtime.h>
#include <hip/hip_bf16.h>
#include <hip/hip_fp16.h>

typedef __attribute__((ext_vector_type(16))) _Float16 v16h;
typedef __attribute__((ext_vector_type(8)))  float    v8f;

#define D  64
#define KH 4
#define TT 2

__device__ __forceinline__ void async_wait0() {
#if __has_builtin(__builtin_amdgcn_s_wait_asynccnt)
  __builtin_amdgcn_s_wait_asynccnt(0);
#else
  asm volatile("s_wait_asynccnt 0x0" ::: "memory");
#endif
}

// Issue one 16B async global->LDS copy. lds_off = low 32 bits of generic LDS
// pointer (== LDS byte offset; aperture is in the high 32 bits per ISA 10.2).
__device__ __forceinline__ void async_copy_b128(unsigned lds_off,
                                                unsigned long long gaddr) {
  asm volatile("global_load_async_to_lds_b128 %0, %1, off"
               :: "v"(lds_off), "v"(gaddr) : "memory");
}

// ---------------------------------------------------------------------------
// Per-node: h_fused = h_prev + h_seq[t]; v = h_fused @ frame_W + b (normalized);
// U = H3 H2 H1 H0 (Householder product), stored row-major f16. Also zero agg.
// One block = one node, 64 threads; thread j owns column j of U.
// ---------------------------------------------------------------------------
__global__ void frame_kernel(const float* __restrict__ hprev,
                             const float* __restrict__ hseq_t,
                             const float* __restrict__ fW,   // D x (KH*D) row-major
                             const float* __restrict__ fb,
                             _Float16* __restrict__ Uh,      // N x D x D f16
                             float* __restrict__ hfused,
                             float* __restrict__ agg) {
  const int n = blockIdx.x;
  const int j = threadIdx.x;   // 0..63
  __shared__ float hf[D];
  __shared__ float vs[KH * D];
  __shared__ float inv[KH];

  float x = hprev[(size_t)n * D + j] + hseq_t[(size_t)n * D + j];
  hf[j] = x;
  hfused[(size_t)n * D + j] = x;
  agg[(size_t)n * D + j] = 0.0f;
  __syncthreads();

  // v = hf @ fW + fb : each thread computes 4 of the 256 outputs
  for (int c = j; c < KH * D; c += D) {
    float acc = fb[c];
    #pragma unroll
    for (int d = 0; d < D; ++d) acc += hf[d] * fW[d * (KH * D) + c];
    vs[c] = acc;
  }
  __syncthreads();

  if (j < KH) {
    float s = 0.f;
    #pragma unroll
    for (int d = 0; d < D; ++d) { float t = vs[j * D + d]; s += t * t; }
    inv[j] = 1.0f / (sqrtf(s) + 1e-8f);
  }
  __syncthreads();
  for (int c = j; c < KH * D; c += D) vs[c] *= inv[c / D];
  __syncthreads();

  // column j of U: c = H3 H2 H1 H0 e_j,  H_i = I - 2 v_i v_i^T
  float col[D];
  #pragma unroll
  for (int d = 0; d < D; ++d) col[d] = (d == j) ? 1.0f : 0.0f;
  #pragma unroll
  for (int i = 0; i < KH; ++i) {
    float s = 0.f;
    #pragma unroll
    for (int d = 0; d < D; ++d) s += vs[i * D + d] * col[d];
    s *= 2.0f;
    #pragma unroll
    for (int d = 0; d < D; ++d) col[d] -= s * vs[i * D + d];
  }
  _Float16* Un = Uh + (size_t)n * (D * D);
  #pragma unroll
  for (int d = 0; d < D; ++d) Un[d * D + j] = (_Float16)col[d];  // coalesced across threads
}

// ---------------------------------------------------------------------------
// Per-edge: Q = U_row * U_col^T via v_wmma_f32_16x16x32_f16 (written to d_out),
// h_trans = U_row (U_col^T h_fused[col]), msg MLP, atomic scatter into agg.
// One block = one edge, 128 threads = 4 waves; each wave does 4 of 16 Q tiles.
// U gathers are L2-resident (U f16 = 82 MB < 192 MB L2); LDS fill uses
// GLOBAL_LOAD_ASYNC_TO_LDS_B128 (no VGPR staging, ASYNCcnt-tracked).
// ---------------------------------------------------------------------------
__global__ void edge_kernel(const int* __restrict__ ei_t, int Eg,
                            const _Float16* __restrict__ Uh,
                            const float* __restrict__ hfused,
                            const float* __restrict__ mW1, const float* __restrict__ mb1,
                            const float* __restrict__ mW2, const float* __restrict__ mb2,
                            float* __restrict__ agg,
                            float* __restrict__ Qout_t /* Eg x D x D f32 */) {
  const int e   = blockIdx.x;
  const int tid = threadIdx.x;         // 0..127
  const int row = ei_t[e];
  const int col = ei_t[Eg + e];

  __shared__ __align__(16) _Float16 sUr[D * D];
  __shared__ __align__(16) _Float16 sUc[D * D];
  __shared__ float shr[D], shc[D], t1[D], htr[D], m1[D];

  {  // async global->LDS fill: 8KB per matrix, 4 x b128 per thread each
    const unsigned lr = (unsigned)(uintptr_t)(&sUr[0]) + (unsigned)tid * 16u;
    const unsigned lc = (unsigned)(uintptr_t)(&sUc[0]) + (unsigned)tid * 16u;
    const unsigned long long gr =
        (unsigned long long)(uintptr_t)(Uh + (size_t)row * (D * D)) + (unsigned)tid * 16u;
    const unsigned long long gc =
        (unsigned long long)(uintptr_t)(Uh + (size_t)col * (D * D)) + (unsigned)tid * 16u;
    #pragma unroll
    for (int i = 0; i < 4; ++i) {
      async_copy_b128(lr + 2048u * i, gr + 2048u * i);
      async_copy_b128(lc + 2048u * i, gc + 2048u * i);
    }
    if (tid < D) shr[tid] = hfused[(size_t)row * D + tid];
    else         shc[tid - D] = hfused[(size_t)col * D + (tid - D)];
    async_wait0();
  }
  __syncthreads();

  // ---- Q tiles: all threads active (EXEC all ones for WMMA) ----
  const int wave = tid >> 5;
  const int lane = tid & 31;
  const int m16  = lane & 15;
  const int hi   = (lane >> 4) & 1;
  float* qb = Qout_t + (size_t)e * (D * D);
  #pragma unroll
  for (int tt = 0; tt < 4; ++tt) {
    const int tile = wave * 4 + tt;
    const int ti = tile >> 2, tj = tile & 3;
    v8f acc = {};
    #pragma unroll
    for (int kc = 0; kc < 2; ++kc) {
      v16h a, b;
      #pragma unroll
      for (int r = 0; r < 8; ++r) {
        // A(m,k) = U_r[16*ti+m][k]; 16-bit A 16x32 layout (ISA 7.12.2)
        const int ka = kc * 32 + ((r >= 4) ? 16 : 0) + hi * 8 + 2 * (r & 3);
        a[2 * r]     = sUr[(ti * 16 + m16) * D + ka];
        a[2 * r + 1] = sUr[(ti * 16 + m16) * D + ka + 1];
        // B(k,n) = U_c[16*tj+n][k]; same K order as A
        const int kb = kc * 32 + ((r >= 4) ? 16 : 0) + hi * 8 + 2 * (r & 3);
        b[2 * r]     = sUc[(tj * 16 + m16) * D + kb];
        b[2 * r + 1] = sUc[(tj * 16 + m16) * D + kb + 1];
      }
      acc = __builtin_amdgcn_wmma_f32_16x16x32_f16(false, a, false, b,
                                                   (short)0, acc, false, false);
    }
    #pragma unroll
    for (int r = 0; r < 8; ++r) {    // C/D 16x16 f32 layout
      const int mq = ti * 16 + r + hi * 8;
      const int nq = tj * 16 + m16;
      qb[mq * D + nq] = acc[r];
    }
  }

  // ---- h_trans = U_r (U_c^T h_col): two D^2 matvecs instead of using Q ----
  if (tid < D) {
    float s = 0.f;
    #pragma unroll
    for (int j2 = 0; j2 < D; ++j2) s += (float)sUc[j2 * D + tid] * shc[j2];
    t1[tid] = s;
  }
  __syncthreads();
  if (tid < D) {
    float s = 0.f;
    #pragma unroll
    for (int k = 0; k < D; ++k) s += (float)sUr[tid * D + k] * t1[k];
    htr[tid] = s;
  }
  __syncthreads();
  // msg = relu([h_row, h_trans] @ W1 + b1) @ W2 + b2 ; scatter into agg[row]
  if (tid < D) {
    float s = mb1[tid];
    #pragma unroll
    for (int d = 0; d < D; ++d) s += shr[d] * mW1[d * D + tid];
    #pragma unroll
    for (int d = 0; d < D; ++d) s += htr[d] * mW1[(D + d) * D + tid];
    m1[tid] = fmaxf(s, 0.f);
  }
  __syncthreads();
  if (tid < D) {
    float s = mb2[tid];
    #pragma unroll
    for (int d = 0; d < D; ++d) s += m1[d] * mW2[d * D + tid];
    atomicAdd(&agg[(size_t)row * D + tid], s);
  }
}

// ---------------------------------------------------------------------------
// Per-node GRU + LayerNorm. One block = one node, 64 threads.
// ---------------------------------------------------------------------------
__global__ void gru_kernel(const float* __restrict__ hfused,
                           const float* __restrict__ agg,
                           const float* __restrict__ Wx,   // D x 3D
                           const float* __restrict__ Wh,   // D x 3D
                           const float* __restrict__ gb,   // 3D
                           const float* __restrict__ gamma,
                           const float* __restrict__ beta,
                           float* __restrict__ hnew,
                           float* __restrict__ out_h_t,
                           float* __restrict__ out_hlast) {
  const int n = blockIdx.x;
  const int i = threadIdx.x;
  __shared__ float ags[D], hcs[D], red[D], red2[D];
  ags[i] = agg[(size_t)n * D + i];
  hcs[i] = hfused[(size_t)n * D + i];
  __syncthreads();
  float gxz = 0, gxr = 0, gxn = 0, ghz = 0, ghr = 0, ghn = 0;
  #pragma unroll 8
  for (int d = 0; d < D; ++d) {
    const float a = ags[d], h = hcs[d];
    const float* wx = Wx + d * 3 * D;
    const float* wh = Wh + d * 3 * D;
    gxz += a * wx[i];           ghz += h * wh[i];
    gxr += a * wx[D + i];       ghr += h * wh[D + i];
    gxn += a * wx[2 * D + i];   ghn += h * wh[2 * D + i];
  }
  const float z  = 1.f / (1.f + expf(-(gxz + ghz + gb[i])));
  const float r  = 1.f / (1.f + expf(-(gxr + ghr + gb[D + i])));
  const float nn = tanhf(gxn + r * ghn + gb[2 * D + i]);
  float hv = (1.f - z) * nn + z * hcs[i];
  // LayerNorm
  red[i] = hv; red2[i] = hv * hv;
  __syncthreads();
  for (int s = D / 2; s > 0; s >>= 1) {
    if (i < s) { red[i] += red[i + s]; red2[i] += red2[i + s]; }
    __syncthreads();
  }
  const float mu  = red[0] * (1.f / D);
  const float var = red2[0] * (1.f / D) - mu * mu;
  const float out = (hv - mu) * rsqrtf(var + 1e-5f) * gamma[i] + beta[i];
  hnew[(size_t)n * D + i]   = out;
  out_h_t[(size_t)n * D + i] = out;
  if (out_hlast) out_hlast[(size_t)n * D + i] = out;
}

// ---------------------------------------------------------------------------
// Per-edge residual: dis = || h_new[row] - U_r (U_c^T h_new[col]) ||^2
// One block = one edge, 64 threads. Async LDS fill; U gathers hit L2.
// ---------------------------------------------------------------------------
__global__ void dis_kernel(const int* __restrict__ ei_t, int Eg,
                           const _Float16* __restrict__ Uh,
                           const float* __restrict__ hnew,
                           float* __restrict__ out_d_t) {
  const int e = blockIdx.x;
  const int i = threadIdx.x;  // 0..63
  const int row = ei_t[e];
  const int col = ei_t[Eg + e];
  __shared__ __align__(16) _Float16 sUr[D * D];
  __shared__ __align__(16) _Float16 sUc[D * D];
  __shared__ float hr[D], hc[D], t1[D], red[D];
  {
    const unsigned lr = (unsigned)(uintptr_t)(&sUr[0]) + (unsigned)i * 16u;
    const unsigned lc = (unsigned)(uintptr_t)(&sUc[0]) + (unsigned)i * 16u;
    const unsigned long long gr =
        (unsigned long long)(uintptr_t)(Uh + (size_t)row * (D * D)) + (unsigned)i * 16u;
    const unsigned long long gc =
        (unsigned long long)(uintptr_t)(Uh + (size_t)col * (D * D)) + (unsigned)i * 16u;
    #pragma unroll
    for (int k = 0; k < 8; ++k) {
      async_copy_b128(lr + 1024u * k, gr + 1024u * k);
      async_copy_b128(lc + 1024u * k, gc + 1024u * k);
    }
  }
  hr[i] = hnew[(size_t)row * D + i];
  hc[i] = hnew[(size_t)col * D + i];
  async_wait0();
  __syncthreads();
  float s = 0.f;
  #pragma unroll
  for (int j = 0; j < D; ++j) s += (float)sUc[j * D + i] * hc[j];
  t1[i] = s;
  __syncthreads();
  float acc = 0.f;
  #pragma unroll
  for (int k = 0; k < D; ++k) acc += (float)sUr[i * D + k] * t1[k];
  const float dv = hr[i] - acc;
  red[i] = dv * dv;
  __syncthreads();
  for (int st = D / 2; st > 0; st >>= 1) {
    if (i < st) red[i] += red[i + st];
    __syncthreads();
  }
  if (i == 0) out_d_t[e] = red[0];
}

// ---------------------------------------------------------------------------
extern "C" void kernel_launch(void* const* d_in, const int* in_sizes, int n_in,
                              void* d_out, int out_size, void* d_ws, size_t ws_size,
                              hipStream_t stream) {
  const float* hseq = (const float*)d_in[0];
  const int*   ei   = (const int*)d_in[1];   // (T,2,E), int32 per harness convention
  const float* fW   = (const float*)d_in[2];
  const float* fb   = (const float*)d_in[3];
  const float* mW1  = (const float*)d_in[4];
  const float* mb1  = (const float*)d_in[5];
  const float* mW2  = (const float*)d_in[6];
  const float* mb2  = (const float*)d_in[7];
  const float* gWx  = (const float*)d_in[8];
  const float* gWh  = (const float*)d_in[9];
  const float* gb   = (const float*)d_in[10];
  const float* lg   = (const float*)d_in[11];
  const float* lb   = (const float*)d_in[12];

  const int N  = in_sizes[0] / (TT * D);
  const int Eg = in_sizes[1] / (2 * TT);

  // outputs concatenated: h_last (N,D), all_h (T,N,D), all_d (T,E), all_Q (T,E,D,D)
  float* out       = (float*)d_out;
  float* out_hlast = out;
  float* out_allh  = out + (size_t)N * D;
  float* out_alld  = out_allh + (size_t)TT * N * D;
  float* out_allq  = out_alld + (size_t)TT * Eg;

  // workspace: hfused | hnew | agg (f32, N*D each) | U (f16, N*D*D)
  float* ws     = (float*)d_ws;
  float* hfused = ws;
  float* hnew   = hfused + (size_t)N * D;
  float* agg    = hnew + (size_t)N * D;
  _Float16* Uh  = (_Float16*)(agg + (size_t)N * D);

  for (int t = 0; t < TT; ++t) {
    const float* hprev  = (t == 0) ? hseq : hnew;
    const float* hseq_t = hseq + (size_t)t * N * D;
    const int*   ei_t   = ei + (size_t)t * 2 * Eg;

    frame_kernel<<<N, D, 0, stream>>>(hprev, hseq_t, fW, fb, Uh, hfused, agg);
    edge_kernel<<<Eg, 2 * D, 0, stream>>>(ei_t, Eg, Uh, hfused, mW1, mb1, mW2, mb2, agg,
                                          out_allq + (size_t)t * Eg * D * D);
    gru_kernel<<<N, D, 0, stream>>>(hfused, agg, gWx, gWh, gb, lg, lb, hnew,
                                    out_allh + (size_t)t * N * D,
                                    (t == TT - 1) ? out_hlast : nullptr);
    dis_kernel<<<Eg, D, 0, stream>>>(ei_t, Eg, Uh, hnew, out_alld + (size_t)t * Eg);
  }
}
